// Qwen3MoeSparseMoeBlock_Grouped_45964740001918
// MI455X (gfx1250) — matrix operand order repaired and checked by reading, not compiled, not run on hardware
//
#include <hip/hip_runtime.h>

typedef __bf16 bf16;
typedef __attribute__((ext_vector_type(16))) __bf16 v16bf;
typedef __attribute__((ext_vector_type(8)))  __bf16 v8bf;
typedef __attribute__((ext_vector_type(4)))  __bf16 v4bf;
typedef __attribute__((ext_vector_type(2)))  __bf16 v2bf;
typedef __attribute__((ext_vector_type(8)))  float  v8f;

#define TTOK 4096   // B*S tokens
#define HD   2048   // hidden dim
#define ID   1408   // intermediate dim
#define NE   8      // experts
// workspace layout (byte offsets)
#define WS_CNT 0                               // 8 ints
#define WS_OFF 64                              // 9 ints
#define WS_TOK 256                             // NE*TTOK ints
#define WS_WTS (WS_TOK + NE*TTOK*4)            // NE*TTOK floats
#define WS_ACT 262400                          // 2*TTOK*ID bf16 (aligned 256)

// ---------------- zero output accumulator + counters ----------------
__global__ void k_zero(float* __restrict__ out, int* __restrict__ cnt) {
  long i = (long)blockIdx.x * blockDim.x + threadIdx.x;
  long n = (long)TTOK * HD;
  for (long j = i; j < n; j += (long)gridDim.x * blockDim.x) out[j] = 0.0f;
  if (i < NE) cnt[i] = 0;
}

// ---------------- router: logits, softmax, top-2, scatter ----------------
__global__ void k_router(const float* __restrict__ x, const float* __restrict__ gw,
                         float* __restrict__ logits, int* __restrict__ cnt,
                         int* __restrict__ tok, float* __restrict__ wts) {
  const int wid  = threadIdx.x >> 5;
  const int lane = threadIdx.x & 31;
  const int t = blockIdx.x * 4 + wid;
  float acc[NE];
#pragma unroll
  for (int e = 0; e < NE; e++) acc[e] = 0.0f;
  for (int h = lane; h < HD; h += 32) {
    float xv = x[(long)t * HD + h];
#pragma unroll
    for (int e = 0; e < NE; e++) acc[e] += xv * gw[e * HD + h];
  }
#pragma unroll
  for (int e = 0; e < NE; e++)
    for (int m = 16; m >= 1; m >>= 1) acc[e] += __shfl_xor(acc[e], m, 32);
  if (lane == 0) {
    float mx = acc[0];
    for (int e = 1; e < NE; e++) mx = fmaxf(mx, acc[e]);
    float p[NE];
    for (int e = 0; e < NE; e++) { p[e] = __expf(acc[e] - mx); logits[t * NE + e] = acc[e]; }
    int e0 = 0; float p0 = p[0];
    for (int e = 1; e < NE; e++) if (p[e] > p0) { p0 = p[e]; e0 = e; }
    int e1 = -1; float p1 = -1.0f;
    for (int e = 0; e < NE; e++) if (e != e0 && p[e] > p1) { p1 = p[e]; e1 = e; }
    float inv = 1.0f / (p0 + p1);   // renormalized top-k (softmax denom cancels)
    int s0 = atomicAdd(&cnt[e0], 1);
    tok[e0 * TTOK + s0] = t; wts[e0 * TTOK + s0] = p0 * inv;
    int s1 = atomicAdd(&cnt[e1], 1);
    tok[e1 * TTOK + s1] = t; wts[e1 * TTOK + s1] = p1 * inv;
  }
}

// ---------------- exclusive scan over 8 counters ----------------
__global__ void k_prefix(const int* __restrict__ cnt, int* __restrict__ off) {
  if (threadIdx.x == 0 && blockIdx.x == 0) {
    int a = 0;
    for (int e = 0; e < NE; e++) { off[e] = a; a += cnt[e]; }
    off[NE] = a;
  }
}

// helper: build v16bf A/B fragment from two aligned 16B LDS chunks
__device__ __forceinline__ v16bf frag16(const bf16* p0, const bf16* p1) {
  v8bf lo = *(const v8bf*)p0;
  v8bf hi = *(const v8bf*)p1;
  return __builtin_shufflevector(lo, hi, 0,1,2,3,4,5,6,7,8,9,10,11,12,13,14,15);
}
__device__ __forceinline__ v2bf pk2(float a, float b) {
  v2bf t; t.x = (bf16)a; t.y = (bf16)b; return t;
}

// ---------------- stage A: act = silu(x@Wg) * (x@Wu), per expert ----------------
// 256 threads = 8 waves; tile 128(M) x 64(N); each wave 32x32 for both g and u.
__launch_bounds__(256)
__global__ void k_gateup(const float* __restrict__ x, const float* __restrict__ wg,
                         const float* __restrict__ wu, const int* __restrict__ cnt,
                         const int* __restrict__ off, const int* __restrict__ tok,
                         bf16* __restrict__ act) {
  const int e  = blockIdx.z;
  const int mt = blockIdx.y;
  const int n_e = cnt[e];
  if (mt * 128 >= n_e) return;              // uniform: EXEC stays all-1s for WMMA
  const int base = off[e];
  const int n0 = blockIdx.x * 64;

  __shared__ bf16 sA[128 * 32];             // [m][k]
  __shared__ bf16 sBg[64 * 32];             // [n][k] (transposed)
  __shared__ bf16 sBu[64 * 32];

  const int tid = threadIdx.x;
  const int wid = tid >> 5, lane = tid & 31;
  const int wm = (wid >> 1) * 32, wn = (wid & 1) * 32;
  const int lr = lane & 15, lh = lane >> 4;

  v8f zero = {};
  v8f ag[2][2], au[2][2];
#pragma unroll
  for (int i = 0; i < 2; i++)
#pragma unroll
    for (int j = 0; j < 2; j++) { ag[i][j] = zero; au[i][j] = zero; }

  // B-tile loader coords: one (k-pair, n-quad) item per thread
  const int kp = tid >> 4;            // 0..15 -> k rows 2kp, 2kp+1
  const int c4b = tid & 15;           // 0..15 -> n quad

  for (int k0 = 0; k0 < HD; k0 += 32) {
    __syncthreads();
    // A tile: 128x32 f32 -> bf16 (gathered token rows); 4 float4 per thread
#pragma unroll
    for (int q = 0; q < 4; q++) {
      int idx = tid * 4 + q;
      int row = idx >> 3, c4 = idx & 7;
      int ml = mt * 128 + row;
      float4 v = {0.f, 0.f, 0.f, 0.f};
      if (ml < n_e) {
        int t = tok[e * TTOK + ml];
        v = *(const float4*)&x[(long)t * HD + k0 + c4 * 4];
      }
      v4bf p; p.x = (bf16)v.x; p.y = (bf16)v.y; p.z = (bf16)v.z; p.w = (bf16)v.w;
      *(v4bf*)&sA[row * 32 + c4 * 4] = p;   // 8B ds_store_b64
    }
    // B tiles: two k-rows x 4 n -> pack (k,k+1) bf16 pairs, b32 stores
    {
      long g0 = ((long)e * HD + (k0 + 2 * kp)) * ID + n0 + c4b * 4;
      long g1 = g0 + ID;
      float4 ga = *(const float4*)&wg[g0];
      float4 gb = *(const float4*)&wg[g1];
      float4 ua = *(const float4*)&wu[g0];
      float4 ub = *(const float4*)&wu[g1];
      if (k0 + 32 < HD) {                    // prefetch next k-step weight rows
        __builtin_prefetch(&wg[g0 + 32 * ID], 0, 0);
        __builtin_prefetch(&wu[g0 + 32 * ID], 0, 0);
      }
      int nb = c4b * 4, kk = 2 * kp;
      *(v2bf*)&sBg[(nb + 0) * 32 + kk] = pk2(ga.x, gb.x);
      *(v2bf*)&sBg[(nb + 1) * 32 + kk] = pk2(ga.y, gb.y);
      *(v2bf*)&sBg[(nb + 2) * 32 + kk] = pk2(ga.z, gb.z);
      *(v2bf*)&sBg[(nb + 3) * 32 + kk] = pk2(ga.w, gb.w);
      *(v2bf*)&sBu[(nb + 0) * 32 + kk] = pk2(ua.x, ub.x);
      *(v2bf*)&sBu[(nb + 1) * 32 + kk] = pk2(ua.y, ub.y);
      *(v2bf*)&sBu[(nb + 2) * 32 + kk] = pk2(ua.z, ub.z);
      *(v2bf*)&sBu[(nb + 3) * 32 + kk] = pk2(ua.w, ub.w);
    }
    __syncthreads();

    v16bf af[2], bg[2], bu[2];
#pragma unroll
    for (int mi = 0; mi < 2; mi++) {
      int row = wm + mi * 16 + lr;
      af[mi] = frag16(&sA[row * 32 + lh * 8], &sA[row * 32 + lh * 8 + 16]);
    }
#pragma unroll
    for (int ni = 0; ni < 2; ni++) {
      int nr = wn + ni * 16 + lr;
      bg[ni] = frag16(&sBg[nr * 32 + lh * 8], &sBg[nr * 32 + lh * 8 + 16]);
      bu[ni] = frag16(&sBu[nr * 32 + lh * 8], &sBu[nr * 32 + lh * 8 + 16]);
    }
#pragma unroll
    for (int mi = 0; mi < 2; mi++)
#pragma unroll
      for (int ni = 0; ni < 2; ni++) {
        ag[mi][ni] = __builtin_amdgcn_wmma_f32_16x16x32_bf16(
            false, af[mi], false, bg[ni], (short)0, ag[mi][ni], false, false);
        au[mi][ni] = __builtin_amdgcn_wmma_f32_16x16x32_bf16(
            false, af[mi], false, bu[ni], (short)0, au[mi][ni], false, false);
      }
  }

  // epilogue: silu(g)*u -> bf16 act rows
#pragma unroll
  for (int mi = 0; mi < 2; mi++)
#pragma unroll
    for (int ni = 0; ni < 2; ni++)
#pragma unroll
      for (int r = 0; r < 8; r++) {
        int ml = mt * 128 + wm + mi * 16 + r + 8 * lh;  // C layout: M=r+8*(lane>=16)
        if (ml < n_e) {
          int col = n0 + wn + ni * 16 + lr;             // N = lane%16
          float g = ag[mi][ni][r], u = au[mi][ni][r];
          float s = g / (1.0f + __expf(-g));
          act[(long)(base + ml) * ID + col] = (bf16)(s * u);
        }
      }
}

// ---------------- stage B: out += weight * (act @ Wd), per expert ----------------
// 256 threads = 8 waves; tile 128(M) x 64(N). A-tile staged with async LDS loads.
__launch_bounds__(256)
__global__ void k_down(const bf16* __restrict__ act, const float* __restrict__ wd,
                       const int* __restrict__ cnt, const int* __restrict__ off,
                       const int* __restrict__ tok, const float* __restrict__ wts,
                       float* __restrict__ out) {
  const int e  = blockIdx.z;
  const int mt = blockIdx.y;
  const int n_e = cnt[e];
  if (mt * 128 >= n_e) return;
  const int base = off[e];
  const int h0 = blockIdx.x * 64;

  __shared__ bf16 sA[128 * 32];   // [m][k]
  __shared__ bf16 sB[64 * 32];    // [h][k] (transposed)

  const int tid = threadIdx.x;
  const int wid = tid >> 5, lane = tid & 31;
  const int wm = (wid >> 1) * 32, wn = (wid & 1) * 32;
  const int lr = lane & 15, lh = lane >> 4;

  v8f zero = {};
  v8f acc[2][2];
#pragma unroll
  for (int i = 0; i < 2; i++)
#pragma unroll
    for (int j = 0; j < 2; j++) acc[i][j] = zero;

  const int kp = tid >> 4;        // B loader: 0..15
  const int c4b = tid & 15;

  for (int k0 = 0; k0 < ID; k0 += 32) {
    __syncthreads();
    // A: act rows already bf16 -> async copy global->LDS (ASYNCcnt path)
#pragma unroll
    for (int q = 0; q < 2; q++) {
      int idx = tid * 2 + q;
      int row = idx >> 2, c8 = idx & 3;
      bf16* lp = &sA[row * 32 + c8 * 8];
      if (mt * 128 + row < n_e) {
        const bf16* gp = &act[(long)(base + mt * 128 + row) * ID + k0 + c8 * 8];
        unsigned lds_off = (unsigned)(unsigned long long)(uintptr_t)lp; // low 32 = LDS byte addr
        unsigned long long ga = (unsigned long long)(uintptr_t)gp;
        asm volatile("global_load_async_to_lds_b128 %0, %1, off"
                     :: "v"(lds_off), "v"(ga) : "memory");
      } else {
        v8bf z = {};
        *(v8bf*)lp = z;                      // zero-fill tail rows
      }
    }
    // B: w_down 32(k) x 64(h) f32 -> bf16 transposed, pair-packed b32 stores
    {
      long g0 = ((long)e * ID + (k0 + 2 * kp)) * HD + h0 + c4b * 4;
      long g1 = g0 + HD;
      float4 da = *(const float4*)&wd[g0];
      float4 db = *(const float4*)&wd[g1];
      if (k0 + 32 < ID) __builtin_prefetch(&wd[g0 + 32 * HD], 0, 0);
      int nb = c4b * 4, kk = 2 * kp;
      *(v2bf*)&sB[(nb + 0) * 32 + kk] = pk2(da.x, db.x);
      *(v2bf*)&sB[(nb + 1) * 32 + kk] = pk2(da.y, db.y);
      *(v2bf*)&sB[(nb + 2) * 32 + kk] = pk2(da.z, db.z);
      *(v2bf*)&sB[(nb + 3) * 32 + kk] = pk2(da.w, db.w);
    }
    asm volatile("s_wait_asynccnt 0x0" ::: "memory");
    __syncthreads();

    v16bf af[2], bfg[2];
#pragma unroll
    for (int mi = 0; mi < 2; mi++) {
      int row = wm + mi * 16 + lr;
      af[mi] = frag16(&sA[row * 32 + lh * 8], &sA[row * 32 + lh * 8 + 16]);
    }
#pragma unroll
    for (int ni = 0; ni < 2; ni++) {
      int nr = wn + ni * 16 + lr;
      bfg[ni] = frag16(&sB[nr * 32 + lh * 8], &sB[nr * 32 + lh * 8 + 16]);
    }
#pragma unroll
    for (int mi = 0; mi < 2; mi++)
#pragma unroll
      for (int ni = 0; ni < 2; ni++)
        acc[mi][ni] = __builtin_amdgcn_wmma_f32_16x16x32_bf16(
            false, af[mi], false, bfg[ni], (short)0, acc[mi][ni], false, false);
  }

  // epilogue: weighted scatter-add (top-2 => 2 adds/token total)
#pragma unroll
  for (int mi = 0; mi < 2; mi++)
#pragma unroll
    for (int ni = 0; ni < 2; ni++)
#pragma unroll
      for (int r = 0; r < 8; r++) {
        int ml = mt * 128 + wm + mi * 16 + r + 8 * lh;
        if (ml < n_e) {
          int   t = tok[e * TTOK + ml];
          float w = wts[e * TTOK + ml];
          int   h = h0 + wn + ni * 16 + lr;
          atomicAdd(&out[(long)t * HD + h], acc[mi][ni][r] * w);
        }
      }
}

extern "C" void kernel_launch(void* const* d_in, const int* in_sizes, int n_in,
                              void* d_out, int out_size, void* d_ws, size_t ws_size,
                              hipStream_t stream) {
  const float* x  = (const float*)d_in[0];   // [T,H]
  const float* gw = (const float*)d_in[1];   // [E,H]
  const float* wg = (const float*)d_in[2];   // [E,H,I]
  const float* wu = (const float*)d_in[3];   // [E,H,I]
  const float* wd = (const float*)d_in[4];   // [E,I,H]
  float* out    = (float*)d_out;             // [T,H]
  float* logits = out + (long)TTOK * HD;     // [T,E]
  char* ws = (char*)d_ws;
  int*   cnt = (int*)(ws + WS_CNT);
  int*   off = (int*)(ws + WS_OFF);
  int*   tok = (int*)(ws + WS_TOK);
  float* wts = (float*)(ws + WS_WTS);
  bf16*  act = (bf16*)(ws + WS_ACT);         // [2T, I] compacted expert rows

  k_zero  <<<1024, 256, 0, stream>>>(out, cnt);
  k_router<<<TTOK / 4, 128, 0, stream>>>(x, gw, logits, cnt, tok, wts);
  k_prefix<<<1, 1, 0, stream>>>(cnt, off);
  dim3 gA(ID / 64, TTOK / 128, NE);          // 22 x 32 x 8 (early-exit on count)
  k_gateup<<<gA, 256, 0, stream>>>(x, wg, wu, cnt, off, tok, act);
  dim3 gB(HD / 64, TTOK / 128, NE);          // 32 x 32 x 8
  k_down  <<<gB, 256, 0, stream>>>(act, wd, cnt, off, tok, wts, out);
}
